// SocialAttention_56822417326383
// MI455X (gfx1250) — matrix operand order repaired and checked by reading, not compiled, not run on hardware
//
#include <hip/hip_runtime.h>
#include <hip/hip_bf16.h>
#include <math.h>

// Problem constants from the reference
#define TT   8
#define NN   1024
#define HID  256
#define FD   64
#define SBS  8
#define EPSF 1e-6f

typedef __attribute__((ext_vector_type(16))) _Float16 v16h;
typedef __attribute__((ext_vector_type(8)))  float    v8f;

__device__ __forceinline__ v8f wmma16(v16h a, v16h b, v8f c) {
    // D(16x16 f32) = A(16x32 f16) x B(32x16 f16) + C
    return __builtin_amdgcn_wmma_f32_16x16x32_f16(
        /*neg_a=*/false, a, /*neg_b=*/false, b,
        /*c_mod=*/(short)0, c, /*reuse_a=*/false, /*reuse_b=*/false);
}

// A fragment (16x32) from LDS row-major [rows][ldk], tile at (m0, k0).
// Lane<16: M=lane, K = 0..15 ; Lane>=16: M=lane-16, K = 16..31
__device__ __forceinline__ v16h load_a_lds(const _Float16* src, int ldk,
                                           int m0, int k0, int lane) {
    int m  = m0 + (lane & 15);
    int ka = (lane < 16) ? 0 : 16;
    const _Float16* r = src + m * ldk + k0;
    v16h a;
#pragma unroll
    for (int e = 0; e < 8; ++e) {
        a[e]     = r[ka + e];
        a[8 + e] = r[ka + 8 + e];
    }
    return a;
}

// B fragment (32x16) from LDS row-major [K][ldn], tile at (k0, n0).
// Lanes 0-15: N=lane, K=k0+0..15 ; lanes 16-31: N=lane-16, K=k0+16..31
__device__ __forceinline__ v16h load_b_lds(const _Float16* src, int ldn,
                                           int k0, int n0, int lane) {
    int n  = n0 + (lane & 15);
    int kb = (lane < 16) ? 0 : 16;
    const _Float16* c = src + (k0 + kb) * ldn + n;
    v16h b;
#pragma unroll
    for (int e = 0; e < 16; ++e) b[e] = c[e * ldn];
    return b;
}

__global__ __launch_bounds__(32)
void social_attention_fused(const float* __restrict__ in_xy,
                            const float* __restrict__ in_dxdy,
                            const float* __restrict__ enc_h,
                            const float* __restrict__ W1, const float* __restrict__ b1,
                            const float* __restrict__ W2, const float* __restrict__ b2,
                            const float* __restrict__ W3, const float* __restrict__ b3,
                            const float* __restrict__ Wa, const float* __restrict__ ba,
                            float* __restrict__ out) {
    __shared__ float     x4s[SBS][4];      // last-timestep [x,y,dx,dy] per agent
    __shared__ float     WhS[16][FD];      // enc_h@Wa+ba (rows 8..15 are pad)
    __shared__ _Float16  h1s[64][32];      // relu(feats@W1+b1)
    __shared__ _Float16  W2h[32 * 64];     // f16 copy of W2
    __shared__ _Float16  h2s[64][64];      // relu(h1@W2+b2)
    __shared__ _Float16  W3h[64 * 64];     // f16 copy of W3
    __shared__ float     fS[64][64];       // h2@W3+b3
    __shared__ float     scoreS[64];
    __shared__ float     attS[64];

    const int lane = threadIdx.x;          // wave32: lane == threadIdx.x
    const int seg  = blockIdx.x;           // 128 segments of 8 agents
    const int base = seg * SBS;

    // Prefetch this segment's enc_h rows (reused by Wh GEMM + final att@enc_h)
    __builtin_prefetch(&enc_h[(size_t)(base + (lane & 7)) * HID], 0, 3);

    // ---- stage 0: last-timestep state x4 = [xy, dxdy] for the 8 agents ----
    {
        int a = lane >> 2, c = lane & 3;
        float v = (c < 2)
                    ? in_xy  [(size_t)(TT - 1) * NN * 2 + (size_t)(base + a) * 2 + c]
                    : in_dxdy[(size_t)(TT - 1) * NN * 2 + (size_t)(base + a) * 2 + (c - 2)];
        x4s[a][c] = v;
    }

    // ---- stage 1: Wh = enc_h[base..base+7] @ Wa + ba  (M padded 8->16) ----
    {
        v8f acc[4] = {};
        int mloc   = lane & 15;
        int magent = base + mloc; if (magent > NN - 1) magent = NN - 1; // pad rows clamp
        const float* rowp = enc_h + (size_t)magent * HID;
        const int ka  = (lane < 16) ? 0 : 16;   // A-fragment K offsets
        const int kbB = (lane < 16) ? 0 : 16;   // B-fragment K offset
        const int nl  = lane & 15;
        for (int kc = 0; kc < HID / 32; ++kc) {
            int k0 = kc * 32;
            v16h a;
#pragma unroll
            for (int e = 0; e < 8; ++e) {
                a[e]     = (_Float16)rowp[k0 + ka + e];
                a[8 + e] = (_Float16)rowp[k0 + ka + 8 + e];
            }
#pragma unroll
            for (int nt = 0; nt < 4; ++nt) {
                v16h b;
#pragma unroll
                for (int e = 0; e < 16; ++e)
                    b[e] = (_Float16)Wa[(size_t)(k0 + kbB + e) * FD + nt * 16 + nl];
                acc[nt] = wmma16(a, b, acc[nt]);
            }
        }
#pragma unroll
        for (int nt = 0; nt < 4; ++nt) {
            int n = nt * 16 + nl;
            float bias = ba[n];
#pragma unroll
            for (int r = 0; r < 8; ++r) {
                int m = (lane < 16) ? r : r + 8;
                WhS[m][n] = acc[nt][r] + bias;
            }
        }
    }

    // ---- stage 2: weights to f16 in LDS ----
    for (int t = lane; t < 32 * 64; t += 32) W2h[t] = (_Float16)W2[t];
    for (int t = lane; t < 64 * 64; t += 32) W3h[t] = (_Float16)W3[t];
    __syncthreads();

    // ---- stage 3: geometric features + layer1 (64 pairs, 2 per lane) ----
    for (int p = lane; p < 64; p += 32) {
        int i = p >> 3, j = p & 7;
        float dpx = x4s[i][0] - x4s[j][0], dpy = x4s[i][1] - x4s[j][1];
        float dvx = x4s[i][2] - x4s[j][2], dvy = x4s[i][3] - x4s[j][3];
        float l2  = sqrtf(dpx * dpx + dpy * dpy);
        float vx  = x4s[i][2], vy = x4s[i][3];
        float vn  = sqrtf(vx * vx + vy * vy);
        float cth = (dpx * vx + dpy * vy) / (l2 * vn + EPSF);
        float dot_pv = dpx * dvx + dpy * dvy;
        float dv_sq  = dvx * dvx + dvy * dvy + EPSF;
        float ttca   = -dot_pv / dv_sq;
        float ex = dpx + ttca * dvx, ey = dpy + ttca * dvy;
        float dca = sqrtf(ex * ex + ey * ey);
#pragma unroll
        for (int o = 0; o < 32; ++o) {
            float h = l2 * W1[o] + cth * W1[32 + o] + dca * W1[64 + o] + b1[o];
            h1s[p][o] = (_Float16)fmaxf(h, 0.0f);
        }
    }
    __syncthreads();

    // ---- stage 4: h2 = relu(h1 @ W2 + b2)   [64x32 @ 32x64] via WMMA ----
    for (int mt = 0; mt < 4; ++mt) {
        v16h a = load_a_lds(&h1s[0][0], 32, mt * 16, 0, lane);
#pragma unroll
        for (int nt = 0; nt < 4; ++nt) {
            v16h b = load_b_lds(W2h, 64, 0, nt * 16, lane);
            v8f c = {};
            c = wmma16(a, b, c);
            int n = nt * 16 + (lane & 15);
            float bias = b2[n];
#pragma unroll
            for (int r = 0; r < 8; ++r) {
                int m = mt * 16 + ((lane < 16) ? r : r + 8);
                h2s[m][n] = (_Float16)fmaxf(c[r] + bias, 0.0f);
            }
        }
    }
    __syncthreads();

    // ---- stage 5: f = h2 @ W3 + b3   [64x64 @ 64x64] via WMMA (K split 2) ----
    for (int mt = 0; mt < 4; ++mt) {
        v16h a0 = load_a_lds(&h2s[0][0], 64, mt * 16, 0,  lane);
        v16h a1 = load_a_lds(&h2s[0][0], 64, mt * 16, 32, lane);
#pragma unroll
        for (int nt = 0; nt < 4; ++nt) {
            v8f c = {};
            c = wmma16(a0, load_b_lds(W3h, 64, 0,  nt * 16, lane), c);
            c = wmma16(a1, load_b_lds(W3h, 64, 32, nt * 16, lane), c);
            int n = nt * 16 + (lane & 15);
            float bias = b3[n];
#pragma unroll
            for (int r = 0; r < 8; ++r) {
                int m = mt * 16 + ((lane < 16) ? r : r + 8);
                fS[m][n] = c[r] + bias;
            }
        }
    }
    __syncthreads();

    // ---- stage 6: sigma[i,j] = f[i,j,:] . Wh[j,:]; diag -> -1000 ----
    for (int p = lane; p < 64; p += 32) {
        int i = p >> 3, j = p & 7;
        float s = 0.0f;
#pragma unroll 8
        for (int n = 0; n < FD; ++n) s += fS[p][n] * WhS[j][n];
        scoreS[p] = (i == j) ? -1000.0f : s;
    }
    __syncthreads();

    // ---- stage 7: softmax over the 8 in-segment entries (one lane per row) ----
    if (lane < 8) {
        float m = -INFINITY;
#pragma unroll
        for (int j = 0; j < 8; ++j) m = fmaxf(m, scoreS[lane * 8 + j]);
        float e[8], den = 0.0f;
#pragma unroll
        for (int j = 0; j < 8; ++j) { e[j] = expf(scoreS[lane * 8 + j] - m); den += e[j]; }
#pragma unroll
        for (int j = 0; j < 8; ++j) attS[lane * 8 + j] = e[j] / den;
    }
    __syncthreads();

    // ---- stage 8: S = att @ enc_h ; keep==true (all segment sizes == 8) ----
    {
        int c0 = lane * (HID / 32); // 8 contiguous columns per lane
        for (int i = 0; i < 8; ++i) {
            float acc[8] = {0.f, 0.f, 0.f, 0.f, 0.f, 0.f, 0.f, 0.f};
#pragma unroll
            for (int j = 0; j < 8; ++j) {
                float w = attS[i * 8 + j];
                const float* er = enc_h + (size_t)(base + j) * HID + c0;
#pragma unroll
                for (int q = 0; q < 8; ++q) acc[q] += w * er[q];
            }
            float* op = out + (size_t)(base + i) * HID + c0;
#pragma unroll
            for (int q = 0; q < 8; ++q) op[q] = acc[q];
        }
    }
}

extern "C" void kernel_launch(void* const* d_in, const int* in_sizes, int n_in,
                              void* d_out, int out_size, void* d_ws, size_t ws_size,
                              hipStream_t stream) {
    (void)in_sizes; (void)n_in; (void)d_ws; (void)ws_size; (void)out_size;
    const float* in_xy   = (const float*)d_in[0];
    const float* in_dxdy = (const float*)d_in[1];
    const float* enc_h   = (const float*)d_in[2];
    // d_in[3] = sub_batches: for N=1024, SB=8 the segments are exactly [8k, 8k+8)
    const float* W1 = (const float*)d_in[4];
    const float* b1 = (const float*)d_in[5];
    const float* W2 = (const float*)d_in[6];
    const float* b2 = (const float*)d_in[7];
    const float* W3 = (const float*)d_in[8];
    const float* b3 = (const float*)d_in[9];
    const float* Wa = (const float*)d_in[10];
    const float* ba = (const float*)d_in[11];
    float* out = (float*)d_out;

    social_attention_fused<<<dim3(NN / SBS), dim3(32), 0, stream>>>(
        in_xy, in_dxdy, enc_h, W1, b1, W2, b2, W3, b3, Wa, ba, out);
}
